// VarianceAdaptor_81192061764167
// MI455X (gfx1250) — matrix operand order repaired and checked by reading, not compiled, not run on hardware
//
#include <hip/hip_runtime.h>
#include <hip/hip_bf16.h>
#include <math.h>

typedef __attribute__((ext_vector_type(16))) _Float16 v16h;
typedef __attribute__((ext_vector_type(8)))  float    v8f;

#define NB 16
#define NL 256
#define NT 2048
#define ND 256
#define NF 256
#define NKTOT 768   // 3 * 256

// ---------------------------------------------------------------------------
// CDNA5 async global->LDS helpers (ASYNCcnt-tracked, see CDNA5 ISA).
// LDS operand: per-lane 32-bit LDS byte address (low 32 bits of generic ptr).
// ---------------------------------------------------------------------------
__device__ __forceinline__ uint32_t lds_off32(const void* p) {
  return (uint32_t)(uintptr_t)p;   // LDS aperture: addr[31:0] = LDS offset
}
__device__ __forceinline__ void async_ld_b64(uint32_t lds, const void* g) {
  asm volatile("global_load_async_to_lds_b64 %0, %1, off"
               :: "v"(lds), "v"(g) : "memory");
}
__device__ __forceinline__ void async_ld_b128(uint32_t lds, const void* g) {
  asm volatile("global_load_async_to_lds_b128 %0, %1, off"
               :: "v"(lds), "v"(g) : "memory");
}
__device__ __forceinline__ void wait_async_le5() {
  asm volatile("s_wait_asynccnt 0x5" ::: "memory");
}
__device__ __forceinline__ void wait_async_le0() {
  asm volatile("s_wait_asynccnt 0x0" ::: "memory");
}

// ---------------------------------------------------------------------------
// Repack conv weights w[k][d][f] (f32, WIO) -> WB[f][k*256+d] (f16, B^T layout)
// ---------------------------------------------------------------------------
__global__ __launch_bounds__(256) void convert_w_kernel(
    const float* __restrict__ W, _Float16* __restrict__ WB) {
  int i = blockIdx.x * 256 + threadIdx.x;          // 0 .. 196607
  int f = i / NKTOT;
  int g = i % NKTOT;                               // g = k*256 + d
  WB[(size_t)f * NKTOT + g] = (_Float16)W[(size_t)g * NF + f];
}

// ---------------------------------------------------------------------------
// x f32 [B,L,256] -> padded f16 [B][L+2][256] (pad rows zeroed)
// ---------------------------------------------------------------------------
__global__ __launch_bounds__(64) void convert_x_kernel(
    const float* __restrict__ X, _Float16* __restrict__ Xp) {
  int m = blockIdx.x;                 // 0 .. NB*(NL+2)-1
  int b = m / (NL + 2), s = m % (NL + 2);
  int d = threadIdx.x * 4;
  _Float16 h0 = (_Float16)0.f, h1 = h0, h2 = h0, h3 = h0;
  if (s >= 1 && s <= NL) {
    float4 v = *(const float4*)(X + ((size_t)b * NL + (s - 1)) * ND + d);
    h0 = (_Float16)v.x; h1 = (_Float16)v.y; h2 = (_Float16)v.z; h3 = (_Float16)v.w;
  }
  _Float16* dst = Xp + (size_t)m * ND + d;
  dst[0] = h0; dst[1] = h1; dst[2] = h2; dst[3] = h3;
}

// ---------------------------------------------------------------------------
// Zero the pad rows (s=0 and s=S+1) of a padded f16 buffer [B][S+2][256]
// ---------------------------------------------------------------------------
__global__ __launch_bounds__(64) void zero_pad_kernel(
    _Float16* __restrict__ Hp, int S) {
  int b = blockIdx.x >> 1;
  int side = blockIdx.x & 1;
  size_t row = (size_t)b * (S + 2) + (side ? (S + 1) : 0);
  unsigned long long z = 0ull;
  *(unsigned long long*)(Hp + row * NF + threadIdx.x * 4) = z;
}

// ---------------------------------------------------------------------------
// conv1d(k=3,pad=1) + bias + LayerNorm + ReLU, WMMA f16->f32.
// Input/output f16, padded layout [B][S+2][256]. Double-buffered LDS tiles
// filled with async global->LDS copies (5 async ops per wave per K-step).
// Workgroup: 256 threads (8 waves). Tile: 32 rows (S) x 256 cols (F).
// ---------------------------------------------------------------------------
__global__ __launch_bounds__(256) void conv_ln_relu_kernel(
    const _Float16* __restrict__ Xp,     // [B][S+2][256] f16 padded
    const _Float16* __restrict__ WB,     // [256][768] f16 (B transposed)
    const float*    __restrict__ bias,   // [256]
    const float*    __restrict__ gamma,  // [256]
    const float*    __restrict__ beta,   // [256]
    _Float16*       __restrict__ Hp,     // [B][S+2][256] f16 padded
    int S) {
  __shared__ __align__(16) _Float16 As[2][32][32];
  __shared__ __align__(16) _Float16 Bs[2][256][32];
  __shared__ __align__(16) float    Cs[32][256];

  const int tid  = threadIdx.x;
  const int lane = tid & 31;
  const int wave = tid >> 5;
  const int wm   = wave >> 2;                 // 0..1 : M sub-tile
  const int wn   = wave & 3;                  // 0..3 : N sub-tile
  const int l16  = lane & 15;
  const int hsel = (lane < 16) ? 0 : 1;

  const int tilesPerB = S >> 5;
  const int b  = blockIdx.x / tilesPerB;
  const int s0 = (blockIdx.x % tilesPerB) << 5;

  // per-thread staging roles
  const int rA   = tid >> 3;                  // A row 0..31
  const int cofA = (tid & 7) << 2;            // A col offset in halfs (8B)
  const uint32_t asBase = lds_off32(&As[0][0][0]);
  const uint32_t bsBase = lds_off32(&Bs[0][0][0]);

  auto stage = [&](int kb2, int buf) {
    const int kbase = kb2 << 5;
    const int ktap  = kbase >> 8;             // conv tap 0..2
    const int dbase = kbase & 255;
    // A tile: 32 rows x 32 halfs; padded input -> no bounds check
    const _Float16* gA =
        Xp + ((size_t)b * (S + 2) + s0 + rA + ktap) * ND + dbase + cofA;
    async_ld_b64(asBase + (uint32_t)(((buf << 5) + rA) * 32 + cofA) * 2, gA);
    // B tile: Bs[buf][tid][0..31] = WB[tid][kbase .. kbase+31]
    const _Float16* gB = WB + (size_t)tid * NKTOT + kbase;
#pragma unroll
    for (int j = 0; j < 4; ++j)
      async_ld_b128(bsBase + (uint32_t)(((buf << 8) + tid) * 32 + j * 8) * 2,
                    gB + j * 8);
  };

  v8f acc[4];
#pragma unroll
  for (int j = 0; j < 4; ++j) acc[j] = (v8f){0.f,0.f,0.f,0.f,0.f,0.f,0.f,0.f};

  stage(0, 0);                                // prologue
  for (int kb = 0; kb < 24; ++kb) {
    const int bufc = kb & 1;
    if (kb < 23) {
      stage(kb + 1, bufc ^ 1);
      wait_async_le5();                       // batch kb landed (in-order)
    } else {
      wait_async_le0();
    }
    __syncthreads();                          // all waves' tiles visible

    union Frag { v16h v; uint4 q[2]; } a, bf;
    {
      int row  = (wm << 4) + l16;
      int koff = hsel << 3;                   // 0 or 8 halfs
      a.q[0] = *(const uint4*)&As[bufc][row][koff];
      a.q[1] = *(const uint4*)&As[bufc][row][16 + koff];
    }
    const int koffB = hsel << 3;              // 0 or 8 halfs
#pragma unroll
    for (int j = 0; j < 4; ++j) {
      int N = (wn << 6) + (j << 4) + l16;
      bf.q[0] = *(const uint4*)&Bs[bufc][N][koffB];
      bf.q[1] = *(const uint4*)&Bs[bufc][N][16 + koffB];
      acc[j] = __builtin_amdgcn_wmma_f32_16x16x32_f16(
          false, a.v, false, bf.v, (short)0, acc[j], false, false);
    }
    __syncthreads();                          // reads done before overwrite
  }

  // ---- spill C + bias to LDS (C/D layout: vgpr r: M=r|r+8, lane: N) ----
#pragma unroll
  for (int j = 0; j < 4; ++j) {
    int col = (wn << 6) + (j << 4) + l16;
    float bcol = bias[col];
#pragma unroll
    for (int r = 0; r < 8; ++r) {
      int row = (wm << 4) + (hsel ? 8 + r : r);
      Cs[row][col] = acc[j][r] + bcol;
    }
  }
  __syncthreads();

  // ---- LayerNorm + ReLU: each wave owns 4 rows, lane covers 8 cols ----
  for (int rr = 0; rr < 4; ++rr) {
    int row = (wave << 2) + rr;
    float4 e0 = *(const float4*)&Cs[row][lane * 8];
    float4 e1 = *(const float4*)&Cs[row][lane * 8 + 4];
    float sum = e0.x + e0.y + e0.z + e0.w + e1.x + e1.y + e1.z + e1.w;
    float sq  = e0.x * e0.x + e0.y * e0.y + e0.z * e0.z + e0.w * e0.w +
                e1.x * e1.x + e1.y * e1.y + e1.z * e1.z + e1.w * e1.w;
#pragma unroll
    for (int off = 16; off > 0; off >>= 1) {
      sum += __shfl_xor(sum, off);
      sq  += __shfl_xor(sq,  off);
    }
    float mu  = sum * (1.0f / 256.0f);
    float var = sq * (1.0f / 256.0f) - mu * mu;
    float rs  = rsqrtf(var + 1e-5f);
    float4 g0 = *(const float4*)(gamma + lane * 8);
    float4 g1 = *(const float4*)(gamma + lane * 8 + 4);
    float4 t0 = *(const float4*)(beta + lane * 8);
    float4 t1 = *(const float4*)(beta + lane * 8 + 4);
    union { _Float16 h[8]; uint4 q; } o;
    o.h[0] = (_Float16)fmaxf(0.f, g0.x * (e0.x - mu) * rs + t0.x);
    o.h[1] = (_Float16)fmaxf(0.f, g0.y * (e0.y - mu) * rs + t0.y);
    o.h[2] = (_Float16)fmaxf(0.f, g0.z * (e0.z - mu) * rs + t0.z);
    o.h[3] = (_Float16)fmaxf(0.f, g0.w * (e0.w - mu) * rs + t0.w);
    o.h[4] = (_Float16)fmaxf(0.f, g1.x * (e1.x - mu) * rs + t1.x);
    o.h[5] = (_Float16)fmaxf(0.f, g1.y * (e1.y - mu) * rs + t1.y);
    o.h[6] = (_Float16)fmaxf(0.f, g1.z * (e1.z - mu) * rs + t1.z);
    o.h[7] = (_Float16)fmaxf(0.f, g1.w * (e1.w - mu) * rs + t1.w);
    _Float16* dst = Hp + ((size_t)b * (S + 2) + s0 + row + 1) * NF + lane * 8;
    *(uint4*)dst = o.q;
  }
}

// ---------------------------------------------------------------------------
// out[row] = dot(Hp[row, :256] (f16 padded), wl) + bl   (one wave per row)
// ---------------------------------------------------------------------------
__global__ __launch_bounds__(256) void rowdot_kernel(
    const _Float16* __restrict__ Hp, const float* __restrict__ wl,
    const float* __restrict__ bl, float* __restrict__ out, int S) {
  int row  = blockIdx.x * 8 + (threadIdx.x >> 5);
  int lane = threadIdx.x & 31;
  int b = row / S, s = row - b * S;
  const _Float16* p = Hp + ((size_t)b * (S + 2) + s + 1) * NF + lane * 8;
  union { uint4 q; _Float16 h[8]; } u;
  u.q = *(const uint4*)p;
  float4 w0 = *(const float4*)(wl + lane * 8);
  float4 w1 = *(const float4*)(wl + lane * 8 + 4);
  float sum = (float)u.h[0] * w0.x + (float)u.h[1] * w0.y +
              (float)u.h[2] * w0.z + (float)u.h[3] * w0.w +
              (float)u.h[4] * w1.x + (float)u.h[5] * w1.y +
              (float)u.h[6] * w1.z + (float)u.h[7] * w1.w;
#pragma unroll
  for (int off = 16; off > 0; off >>= 1) sum += __shfl_xor(sum, off);
  if (lane == 0) out[row] = sum + bl[0];
}

// ---------------------------------------------------------------------------
// Length regulator: per-batch prefix sum of durations + binary-search map
// ---------------------------------------------------------------------------
__global__ __launch_bounds__(256) void build_idx_kernel(
    const int* __restrict__ dur, int* __restrict__ idxmap) {
  __shared__ int s[256];
  int b = blockIdx.x, tid = threadIdx.x;
  s[tid] = dur[b * NL + tid];
  __syncthreads();
  for (int off = 1; off < 256; off <<= 1) {
    int v = s[tid];
    int a = (tid >= off) ? s[tid - off] : 0;
    __syncthreads();
    s[tid] = v + a;
    __syncthreads();
  }
  int total = s[255];
  for (int i = 0; i < 8; ++i) {
    int t = tid + i * 256;
    int res = -1;
    if (t < total) {
      int lo = 0, hi = 255;
      while (lo < hi) { int mid = (lo + hi) >> 1; if (s[mid] > t) hi = mid; else lo = mid + 1; }
      res = lo;
    }
    idxmap[b * NT + t] = res;
  }
}

// gather: xe f32 (for output path) + padded f16 copy (for conv input)
__global__ __launch_bounds__(64) void gather_kernel(
    const float* __restrict__ X, const int* __restrict__ idxmap,
    float* __restrict__ XE, _Float16* __restrict__ XE16p) {
  int bt = blockIdx.x;
  int b = bt / NT, t = bt % NT;
  int l = idxmap[bt];
  int d = threadIdx.x * 4;
  float4 v = make_float4(0.f, 0.f, 0.f, 0.f);
  if (l >= 0) v = *(const float4*)(X + ((size_t)b * NL + l) * ND + d);
  *(float4*)(XE + (size_t)bt * ND + d) = v;
  _Float16* h = XE16p + ((size_t)b * (NT + 2) + t + 1) * ND + d;
  h[0] = (_Float16)v.x; h[1] = (_Float16)v.y;
  h[2] = (_Float16)v.z; h[3] = (_Float16)v.w;
}

// ---------------------------------------------------------------------------
// OUT = IN + emb[searchsorted(linspace(0,1,256), v, left)], v per mode.
// Optionally also writes a padded f16 copy (next conv's input).
// ---------------------------------------------------------------------------
__global__ __launch_bounds__(64) void add_emb_kernel(
    const float* __restrict__ IN, const float* __restrict__ target,
    const float* __restrict__ emb, float* __restrict__ OUT,
    _Float16* __restrict__ OUT16p, int mode) {
  int bt = blockIdx.x;
  int b = bt / NT, t = bt % NT;
  float tv = target[bt];
  float v = (mode == 0) ? (log1pf(tv) * (1.0f / 6.6858613918f))   // 1/log1p(800)
                        : (tv * 0.01f);
  v = fminf(fmaxf(v, 0.0f), 1.0f);
  int idx = (int)ceilf(v * 255.0f);
  idx = idx < 0 ? 0 : (idx > 255 ? 255 : idx);
  int d = threadIdx.x * 4;
  float4 a = *(const float4*)(IN + (size_t)bt * ND + d);
  float4 e = *(const float4*)(emb + (size_t)idx * ND + d);
  a.x += e.x; a.y += e.y; a.z += e.z; a.w += e.w;
  *(float4*)(OUT + (size_t)bt * ND + d) = a;
  if (OUT16p) {
    _Float16* h = OUT16p + ((size_t)b * (NT + 2) + t + 1) * ND + d;
    h[0] = (_Float16)a.x; h[1] = (_Float16)a.y;
    h[2] = (_Float16)a.z; h[3] = (_Float16)a.w;
  }
}

// ---------------------------------------------------------------------------
extern "C" void kernel_launch(void* const* d_in, const int* in_sizes, int n_in,
                              void* d_out, int out_size, void* d_ws, size_t ws_size,
                              hipStream_t stream) {
  (void)in_sizes; (void)n_in; (void)out_size; (void)ws_size;

  const float* x        = (const float*)d_in[0];
  const int*   length_t = (const int*)d_in[1];
  const float* pitch_t  = (const float*)d_in[2];
  const float* energy_t = (const float*)d_in[3];
  const int DUR = 5, PIT = 15, ENE = 25;   // w1,b1,g1,be1,w2,b2,g2,be2,wl,bl
  const float* pitch_emb  = (const float*)d_in[35];
  const float* energy_emb = (const float*)d_in[36];

  // Workspace layout (bytes)
  char* ws = (char*)d_ws;
  const size_t WBSZ   = (size_t)NF * NKTOT * sizeof(_Float16);     // 393216
  const size_t PAD_L  = (size_t)NB * (NL + 2) * ND * sizeof(_Float16);
  const size_t PAD_T  = (size_t)NB * (NT + 2) * ND * sizeof(_Float16);
  size_t off = 0;
  _Float16* wb[6];
  for (int i = 0; i < 6; ++i) { wb[i] = (_Float16*)(ws + off); off += WBSZ; }
  int*      idxmap = (int*)(ws + off);      off += (size_t)NB * NT * 4;
  float*    buf0   = (float*)(ws + off);    off += (size_t)NB * NT * NF * 4; // xe f32
  _Float16* x16p   = (_Float16*)(ws + off); off += PAD_L;
  _Float16* hd1    = (_Float16*)(ws + off); off += PAD_L;
  _Float16* hd2    = (_Float16*)(ws + off); off += PAD_L;
  _Float16* xe16p  = (_Float16*)(ws + off); off += PAD_T;
  _Float16* h1p    = (_Float16*)(ws + off); off += PAD_T;
  _Float16* h2p    = (_Float16*)(ws + off); off += PAD_T;
  _Float16* xe2p   = (_Float16*)(ws + off); off += PAD_T;

  float* out_xe  = (float*)d_out;
  float* out_dur = out_xe + (size_t)NB * NT * ND;
  float* out_pit = out_dur + NB * NL;
  float* out_ene = out_pit + NB * NT;

  // 0) repack weights to f16 B^T layout; convert/pad x
  const int wsrc[6] = {DUR + 0, DUR + 4, PIT + 0, PIT + 4, ENE + 0, ENE + 4};
  for (int i = 0; i < 6; ++i)
    convert_w_kernel<<<NKTOT * NF / 256, 256, 0, stream>>>(
        (const float*)d_in[wsrc[i]], wb[i]);
  convert_x_kernel<<<NB * (NL + 2), 64, 0, stream>>>(x, x16p);
  zero_pad_kernel<<<NB * 2, 64, 0, stream>>>(hd1, NL);
  zero_pad_kernel<<<NB * 2, 64, 0, stream>>>(xe16p, NT);
  zero_pad_kernel<<<NB * 2, 64, 0, stream>>>(h1p, NT);
  zero_pad_kernel<<<NB * 2, 64, 0, stream>>>(xe2p, NT);

  // 1) duration predictor (S = L = 256)
  conv_ln_relu_kernel<<<NB * (NL / 32), 256, 0, stream>>>(
      x16p, wb[0], (const float*)d_in[DUR + 1], (const float*)d_in[DUR + 2],
      (const float*)d_in[DUR + 3], hd1, NL);
  conv_ln_relu_kernel<<<NB * (NL / 32), 256, 0, stream>>>(
      hd1, wb[1], (const float*)d_in[DUR + 5], (const float*)d_in[DUR + 6],
      (const float*)d_in[DUR + 7], hd2, NL);
  rowdot_kernel<<<NB * NL / 8, 256, 0, stream>>>(
      hd2, (const float*)d_in[DUR + 8], (const float*)d_in[DUR + 9],
      out_dur, NL);

  // 2) length regulate -> xe (f32 buf0 + padded f16 xe16p)
  build_idx_kernel<<<NB, 256, 0, stream>>>(length_t, idxmap);
  gather_kernel<<<NB * NT, 64, 0, stream>>>(x, idxmap, buf0, xe16p);

  // 3) pitch predictor on xe (S = T = 2048)
  conv_ln_relu_kernel<<<NB * (NT / 32), 256, 0, stream>>>(
      xe16p, wb[2], (const float*)d_in[PIT + 1], (const float*)d_in[PIT + 2],
      (const float*)d_in[PIT + 3], h1p, NT);
  conv_ln_relu_kernel<<<NB * (NT / 32), 256, 0, stream>>>(
      h1p, wb[3], (const float*)d_in[PIT + 5], (const float*)d_in[PIT + 6],
      (const float*)d_in[PIT + 7], h2p, NT);
  rowdot_kernel<<<NB * NT / 8, 256, 0, stream>>>(
      h2p, (const float*)d_in[PIT + 8], (const float*)d_in[PIT + 9],
      out_pit, NT);

  // 4) xe2 = xe + pitch_emb[pidx]; f32 to d_out, f16 padded copy to xe2p
  add_emb_kernel<<<NB * NT, 64, 0, stream>>>(
      buf0, pitch_t, pitch_emb, out_xe, xe2p, 0);

  // 5) energy predictor on xe2 (f16 padded)
  conv_ln_relu_kernel<<<NB * (NT / 32), 256, 0, stream>>>(
      xe2p, wb[4], (const float*)d_in[ENE + 1], (const float*)d_in[ENE + 2],
      (const float*)d_in[ENE + 3], h1p, NT);
  conv_ln_relu_kernel<<<NB * (NT / 32), 256, 0, stream>>>(
      h1p, wb[5], (const float*)d_in[ENE + 5], (const float*)d_in[ENE + 6],
      (const float*)d_in[ENE + 7], h2p, NT);
  rowdot_kernel<<<NB * NT / 8, 256, 0, stream>>>(
      h2p, (const float*)d_in[ENE + 8], (const float*)d_in[ENE + 9],
      out_ene, NT);

  // 6) xe3 = xe2 + energy_emb[eidx] (element-wise read-then-write, same call)
  add_emb_kernel<<<NB * NT, 64, 0, stream>>>(
      out_xe, energy_t, energy_emb, out_xe, nullptr, 1);
}
